// MoE_20796231647727
// MI455X (gfx1250) — matrix operand order repaired and checked by reading, not compiled, not run on hardware
//
#include <hip/hip_runtime.h>
#include <hip/hip_bf16.h>

// ---------------- types ----------------
typedef __attribute__((ext_vector_type(16))) __bf16 bf16x16;
typedef __attribute__((ext_vector_type(8)))  float  f32x8;
typedef unsigned int u32x4 __attribute__((ext_vector_type(4)));
typedef int          i32x8 __attribute__((ext_vector_type(8)));
typedef int          i32x4 __attribute__((ext_vector_type(4)));

#define NTOK   8192
#define CDIM   1024
#define HDIM   2048
#define NEXP   8
#define TMR    128                   // rows per GEMM tile (routing pad unit)
#define RMAX   (2*NTOK + NEXP*TMR)   // 17408 padded routing rows
#define NTILES (RMAX/TMR)            // 136 row tiles

// ---------------- helpers ----------------
__device__ __forceinline__ unsigned short f2bf(float f) {
    unsigned int u = __float_as_uint(f);
    unsigned int r = u + 0x7FFFu + ((u >> 16) & 1u);
    return (unsigned short)(r >> 16);
}

// A-fragment (16x32, 16-bit): lane half lh -> K {lh*16..+15} contiguous
__device__ __forceinline__ bf16x16 ldfragA(const unsigned short* p) {
    union { uint4 q[2]; bf16x16 v; } u;
    u.q[0] = *(const uint4*)(p);
    u.q[1] = *(const uint4*)(p + 8);
    return u.v;
}
// B-fragment (32x16, 16-bit): lane half lh -> K {lh*16..+15} contiguous
__device__ __forceinline__ bf16x16 ldfragB(const unsigned short* p) {
    union { uint4 q[2]; bf16x16 v; } u;
    u.q[0] = *(const uint4*)(p);
    u.q[1] = *(const uint4*)(p + 8);
    return u.v;
}
__device__ __forceinline__ f32x8 wmma_bf16(bf16x16 a, bf16x16 b, f32x8 c) {
    return __builtin_amdgcn_wmma_f32_16x16x32_bf16(false, a, false, b,
                                                   (short)0, c, false, false);
}

// LDS byte address (low 32 bits of the flat address = LDS offset)
__device__ __forceinline__ unsigned lds_addr_of(const void* p) {
    return (unsigned)(unsigned long long)p;
}

// Tensor Data Mover: 2D tile load global->LDS, bf16 elements, row width 64
// elements (128B = 32 DWORDs) padded with 4 DWORDs (-> 72-ushort LDS pitch).
__device__ __forceinline__ void tdm_load_2d(unsigned lds_byte, const void* gptr,
                                            unsigned tile_h,
                                            unsigned long long stride_elems,
                                            unsigned tensor_w, unsigned tensor_h) {
    unsigned long long ga = (unsigned long long)gptr;
    u32x4 g0; i32x8 g1; i32x4 g2, g3; i32x8 g4;
    g0[0] = 1u;                                           // count=1 (valid), user mode
    g0[1] = lds_byte;                                     // lds_addr
    g0[2] = (unsigned)ga;                                 // global_addr[31:0]
    g0[3] = (unsigned)((ga >> 32) & 0x01FFFFFFu) | (2u << 30);  // addr[56:32], type=2
    // data_size=1(2B), pad_enable, pad_interval=32 DW (code 4), pad_amount=4 DW (code 3)
    g1[0] = (int)((1u << 16) | (1u << 20) | (4u << 22) | (3u << 25));
    g1[1] = (int)((tensor_w & 0xFFFFu) << 16);            // tensor_dim0[15:0]
    g1[2] = (int)((tensor_w >> 16) | ((tensor_h & 0xFFFFu) << 16)); // td0 hi | td1 lo
    g1[3] = (int)((tensor_h >> 16) | (64u << 16));        // td1 hi | tile_dim0=64
    g1[4] = (int)(tile_h & 0xFFFFu);                      // tile_dim1, tile_dim2=0
    g1[5] = (int)(unsigned)(stride_elems & 0xFFFFFFFFull);// tensor_dim0_stride lo
    g1[6] = (int)(unsigned)((stride_elems >> 32) & 0xFFFFull); // stride hi, td1_stride lo=0
    g1[7] = 0;
    g2[0] = 0; g2[1] = 0; g2[2] = 0; g2[3] = 0;
    g3[0] = 0; g3[1] = 0; g3[2] = 0; g3[3] = 0;
    g4[0] = 0; g4[1] = 0; g4[2] = 0; g4[3] = 0;
    g4[4] = 0; g4[5] = 0; g4[6] = 0; g4[7] = 0;
    __builtin_amdgcn_tensor_load_to_lds(g0, g1, g2, g3, g4, 0);
}

// ---------------- conversion kernels ----------------
__global__ __launch_bounds__(256) void k_convX(const float* __restrict__ x,
                                               unsigned short* __restrict__ xb) {
    size_t i = (size_t)blockIdx.x * 256 + threadIdx.x;
    xb[i] = f2bf(x[i]);
}

// transpose-convert: in [R][Cc] fp32 -> out [Cc][R] bf16, per expert (blockIdx.z)
__global__ __launch_bounds__(256) void k_convT(const float* __restrict__ in,
                                               unsigned short* __restrict__ out,
                                               int R, int Cc) {
    __shared__ float tile[64][65];
    int e = blockIdx.z;
    const float* src = in + (size_t)e * R * Cc;
    unsigned short* dst = out + (size_t)e * R * Cc;
    int r0 = blockIdx.x * 64, c0 = blockIdx.y * 64;
    int tx = threadIdx.x & 63, ty = threadIdx.x >> 6;  // ty 0..3
    #pragma unroll
    for (int i = 0; i < 16; i++) {
        int r = ty + i * 4;
        tile[r][tx] = src[(size_t)(r0 + r) * Cc + c0 + tx];
    }
    __syncthreads();
    #pragma unroll
    for (int i = 0; i < 16; i++) {
        int cc = ty + i * 4;
        dst[(size_t)(c0 + cc) * R + r0 + tx] = f2bf(tile[tx][cc]);
    }
}

// ---------------- gating ----------------
__global__ __launch_bounds__(256) void k_gate(const float* __restrict__ x,
                                              const float* __restrict__ Wg,
                                              int* __restrict__ eidx,
                                              float* __restrict__ wgt,
                                              int* __restrict__ cnt,
                                              float* __restrict__ partials) {
    __shared__ float bl[8][16];
    int wid = threadIdx.x >> 5, lane = threadIdx.x & 31;
    int t = blockIdx.x * 8 + wid;
    float g[8] = {0, 0, 0, 0, 0, 0, 0, 0};
    for (int c = lane; c < CDIM; c += 32) {
        float xv = x[(size_t)t * CDIM + c];
        const float4* w = (const float4*)(Wg + (size_t)c * 8);
        float4 w0 = w[0], w1 = w[1];
        g[0] += xv * w0.x; g[1] += xv * w0.y; g[2] += xv * w0.z; g[3] += xv * w0.w;
        g[4] += xv * w1.x; g[5] += xv * w1.y; g[6] += xv * w1.z; g[7] += xv * w1.w;
    }
    #pragma unroll
    for (int e = 0; e < 8; e++)
        #pragma unroll
        for (int off = 16; off >= 1; off >>= 1)
            g[e] += __shfl_xor(g[e], off, 32);
    if (lane == 0) {
        int i0 = 0; float m0 = g[0];
        #pragma unroll
        for (int e = 1; e < 8; e++) if (g[e] > m0) { m0 = g[e]; i0 = e; }
        int i1 = (i0 == 0) ? 1 : 0; float m1 = g[i1];
        #pragma unroll
        for (int e = 0; e < 8; e++)
            if (e != i0 && g[e] > m1) { m1 = g[e]; i1 = e; }
        float e1 = __expf(m1 - m0);
        float s2 = 1.0f + e1;
        eidx[2 * t] = i0; eidx[2 * t + 1] = i1;
        wgt[2 * t] = 1.0f / s2; wgt[2 * t + 1] = e1 / s2;
        atomicAdd(&cnt[i0], 1);
        atomicAdd(&cnt[i1], 1);
        float p[8], ps = 0.0f;
        #pragma unroll
        for (int e = 0; e < 8; e++) { p[e] = __expf(g[e] - m0); ps += p[e]; }
        #pragma unroll
        for (int e = 0; e < 8; e++) {
            bl[wid][e] = (e == i0) ? 1.0f : 0.0f;
            bl[wid][8 + e] = p[e] / ps;
        }
    }
    __syncthreads();
    if (threadIdx.x < 16) {
        float s = 0.0f;
        #pragma unroll
        for (int w = 0; w < 8; w++) s += bl[w][threadIdx.x];
        partials[(size_t)blockIdx.x * 16 + threadIdx.x] = s;
    }
}

__global__ void k_offs(const int* __restrict__ cnt, int* __restrict__ offs) {
    if (threadIdx.x == 0) {
        int o = 0;
        for (int e = 0; e < NEXP; e++) {
            offs[e] = o;
            o += (cnt[e] + TMR - 1) & ~(TMR - 1);
        }
        offs[NEXP] = o;
    }
}

__global__ void k_reduce(const float* __restrict__ partials, float* __restrict__ aux) {
    __shared__ float red[16];
    int i = threadIdx.x;
    if (i < 16) {
        float s = 0.0f;
        for (int b = 0; b < NTOK / 8; b++) s += partials[(size_t)b * 16 + i];
        red[i] = s;
    }
    __syncthreads();
    if (i == 0) {
        const float invN = 1.0f / (float)NTOK;
        float s = 0.0f;
        for (int e = 0; e < 8; e++) s += (red[e] * invN) * (red[8 + e] * invN);
        aux[0] = 0.01f * 8.0f * s;  // AUX_WEIGHT * E
    }
}

__global__ __launch_bounds__(256) void k_padinit(int* __restrict__ rowmap,
                                                 float* __restrict__ roww) {
    int i = blockIdx.x * 256 + threadIdx.x;
    rowmap[i] = -1;
    roww[i] = 0.0f;
}

__global__ __launch_bounds__(256) void k_route(const int* __restrict__ eidx,
                                               const float* __restrict__ wgt,
                                               int* __restrict__ fill,
                                               const int* __restrict__ offs,
                                               int* __restrict__ rowmap,
                                               float* __restrict__ roww) {
    int t = blockIdx.x * 256 + threadIdx.x;
    #pragma unroll
    for (int k = 0; k < 2; k++) {
        int e = eidx[2 * t + k];
        int p = atomicAdd(&fill[e], 1);
        int row = offs[e] + p;
        rowmap[row] = t;
        roww[row] = wgt[2 * t + k];
    }
}

// ---------------- GEMM1: h = silu(X@W1) * (X@W3), 128x64 tile ----------------
__global__ __launch_bounds__(256) void k_gemm1(const unsigned short* __restrict__ xb,
                                               const unsigned short* __restrict__ W1t,
                                               const unsigned short* __restrict__ W3t,
                                               const int* __restrict__ rowmap,
                                               const int* __restrict__ offs,
                                               unsigned short* __restrict__ hbuf) {
    __shared__ unsigned short Xs[128][72];
    __shared__ unsigned short As[64][72];
    __shared__ unsigned short Bs[64][72];
    int rowStart = blockIdx.x * TMR;
    if (rowStart >= offs[NEXP]) return;
    int e = 0;
    #pragma unroll
    for (int i = 0; i < NEXP - 1; i++) if (rowStart >= offs[i + 1]) e = i + 1;
    int ncol0 = blockIdx.y * 64;
    const unsigned short* Wa = W1t + (size_t)e * HDIM * CDIM + (size_t)ncol0 * CDIM;
    const unsigned short* Wb = W3t + (size_t)e * HDIM * CDIM + (size_t)ncol0 * CDIM;

    int tid = threadIdx.x;
    int lane = tid & 31, wid = tid >> 5;
    int wm = wid & 1, wn = wid >> 1;          // 64-row half, 16-col group
    int lr = lane & 15, lh = lane >> 4;
    int ldr = tid >> 3;                       // 0..31
    int seg = (tid & 7) * 8;                  // 0..56

    int tok[4];
    #pragma unroll
    for (int r = 0; r < 4; r++) tok[r] = rowmap[rowStart + ldr + r * 32];

    unsigned ldsA = lds_addr_of(&As[0][0]);
    unsigned ldsB = lds_addr_of(&Bs[0][0]);

    f32x8 accA[4] = {}, accB[4] = {};
    for (int ko = 0; ko < CDIM; ko += 64) {
        if (wid == 0) {  // TDM: stream both weight tiles into LDS
            tdm_load_2d(ldsA, Wa + ko, 64, CDIM, CDIM, HDIM);
            tdm_load_2d(ldsB, Wb + ko, 64, CDIM, CDIM, HDIM);
            __builtin_amdgcn_s_wait_tensorcnt(0);
        }
        #pragma unroll
        for (int r = 0; r < 4; r++) {
            uint4 v = make_uint4(0, 0, 0, 0);
            if (tok[r] >= 0) v = *(const uint4*)(xb + (size_t)tok[r] * CDIM + ko + seg);
            *(uint4*)&Xs[ldr + r * 32][seg] = v;
        }
        __syncthreads();
        #pragma unroll
        for (int kk = 0; kk < 64; kk += 32) {
            bf16x16 ba = ldfragB(&As[wn * 16 + lr][kk + lh * 16]);
            bf16x16 bb = ldfragB(&Bs[wn * 16 + lr][kk + lh * 16]);
            #pragma unroll
            for (int s = 0; s < 4; s++) {
                bf16x16 a = ldfragA(&Xs[wm * 64 + s * 16 + lr][kk + lh * 16]);
                accA[s] = wmma_bf16(a, ba, accA[s]);
                accB[s] = wmma_bf16(a, bb, accB[s]);
            }
        }
        __syncthreads();
    }
    int col = ncol0 + wn * 16 + lr;
    #pragma unroll
    for (int s = 0; s < 4; s++) {
        #pragma unroll
        for (int j = 0; j < 8; j++) {
            int gr = rowStart + wm * 64 + s * 16 + lh * 8 + j;
            float a = accA[s][j], b = accB[s][j];
            float hv = (a / (1.0f + __expf(-a))) * b;
            hbuf[(size_t)gr * HDIM + col] = f2bf(hv);
        }
    }
}

// ---------------- GEMM2: out[token] += w * (h @ W2), 128x128 tile ----------------
__global__ __launch_bounds__(256) void k_gemm2(const unsigned short* __restrict__ hbuf,
                                               const unsigned short* __restrict__ W2t,
                                               const int* __restrict__ rowmap,
                                               const float* __restrict__ roww,
                                               const int* __restrict__ offs,
                                               float* __restrict__ out) {
    __shared__ unsigned short Hs[128][72];
    __shared__ unsigned short Ws[128][72];
    int rowStart = blockIdx.x * TMR;
    if (rowStart >= offs[NEXP]) return;
    int e = 0;
    #pragma unroll
    for (int i = 0; i < NEXP - 1; i++) if (rowStart >= offs[i + 1]) e = i + 1;
    int ccol0 = blockIdx.y * 128;
    const unsigned short* Wp = W2t + (size_t)e * CDIM * HDIM + (size_t)ccol0 * HDIM;

    int tid = threadIdx.x;
    int lane = tid & 31, wid = tid >> 5;
    int wm = wid & 1, wn = wid >> 1;          // 64-row half, 32-col group
    int lr = lane & 15, lh = lane >> 4;

    unsigned ldsH = lds_addr_of(&Hs[0][0]);
    unsigned ldsW = lds_addr_of(&Ws[0][0]);

    f32x8 acc[4][2] = {};
    for (int ko = 0; ko < HDIM; ko += 64) {
        if (wid == 0) {  // TDM: both input tiles streamed by the tensor DMA
            tdm_load_2d(ldsH, hbuf + (size_t)rowStart * HDIM + ko, 128, HDIM, HDIM, RMAX);
            tdm_load_2d(ldsW, Wp + ko, 128, HDIM, HDIM, CDIM);
            __builtin_amdgcn_s_wait_tensorcnt(0);
        }
        __syncthreads();
        #pragma unroll
        for (int kk = 0; kk < 64; kk += 32) {
            bf16x16 b0 = ldfragB(&Ws[wn * 32 + lr][kk + lh * 16]);
            bf16x16 b1 = ldfragB(&Ws[wn * 32 + 16 + lr][kk + lh * 16]);
            #pragma unroll
            for (int s = 0; s < 4; s++) {
                bf16x16 a = ldfragA(&Hs[wm * 64 + s * 16 + lr][kk + lh * 16]);
                acc[s][0] = wmma_bf16(a, b0, acc[s][0]);
                acc[s][1] = wmma_bf16(a, b1, acc[s][1]);
            }
        }
        __syncthreads();
    }
    #pragma unroll
    for (int s = 0; s < 4; s++) {
        #pragma unroll
        for (int n2 = 0; n2 < 2; n2++) {
            int col = ccol0 + wn * 32 + n2 * 16 + lr;
            #pragma unroll
            for (int j = 0; j < 8; j++) {
                int gr = rowStart + wm * 64 + s * 16 + lh * 8 + j;
                int t = rowmap[gr];
                if (t >= 0) {
                    float val = acc[s][n2][j] * roww[gr];
                    __hip_atomic_fetch_add(&out[(size_t)t * CDIM + col], val,
                                           __ATOMIC_RELAXED, __HIP_MEMORY_SCOPE_AGENT);
                }
            }
        }
    }
}

// ---------------- launch ----------------
extern "C" void kernel_launch(void* const* d_in, const int* in_sizes, int n_in,
                              void* d_out, int out_size, void* d_ws, size_t ws_size,
                              hipStream_t stream) {
    (void)in_sizes; (void)n_in; (void)out_size; (void)ws_size;
    const float* x  = (const float*)d_in[0];
    const float* Wg = (const float*)d_in[1];
    const float* W1 = (const float*)d_in[2];
    const float* W2 = (const float*)d_in[3];
    const float* W3 = (const float*)d_in[4];
    float* out = (float*)d_out;

    char* ws = (char*)d_ws;
    size_t off = 0;
    auto alloc = [&](size_t bytes) -> void* {
        void* p = ws + off;
        off += (bytes + 255) & ~(size_t)255;
        return p;
    };
    unsigned short* xb   = (unsigned short*)alloc((size_t)NTOK * CDIM * 2);
    unsigned short* W1t  = (unsigned short*)alloc((size_t)NEXP * HDIM * CDIM * 2);
    unsigned short* W3t  = (unsigned short*)alloc((size_t)NEXP * HDIM * CDIM * 2);
    unsigned short* W2t  = (unsigned short*)alloc((size_t)NEXP * HDIM * CDIM * 2);
    unsigned short* hbuf = (unsigned short*)alloc((size_t)RMAX * HDIM * 2);
    int*   rowmap   = (int*)alloc((size_t)RMAX * 4);
    float* roww     = (float*)alloc((size_t)RMAX * 4);
    int*   eidx     = (int*)alloc((size_t)NTOK * 2 * 4);
    float* wgt      = (float*)alloc((size_t)NTOK * 2 * 4);
    int*   cnt      = (int*)alloc(16 * 4);           // cnt[8] + fill[8]
    int*   fill     = cnt + 8;
    int*   offs     = (int*)alloc(16 * 4);
    float* partials = (float*)alloc((size_t)(NTOK / 8) * 16 * 4);

    (void)hipMemsetAsync(cnt, 0, 16 * 4, stream);
    (void)hipMemsetAsync(out, 0, (size_t)NTOK * CDIM * 4, stream);

    k_convX<<<(NTOK * CDIM) / 256, 256, 0, stream>>>(x, xb);
    k_convT<<<dim3(CDIM / 64, HDIM / 64, NEXP), 256, 0, stream>>>(W1, W1t, CDIM, HDIM);
    k_convT<<<dim3(CDIM / 64, HDIM / 64, NEXP), 256, 0, stream>>>(W3, W3t, CDIM, HDIM);
    k_convT<<<dim3(HDIM / 64, CDIM / 64, NEXP), 256, 0, stream>>>(W2, W2t, HDIM, CDIM);

    k_gate<<<NTOK / 8, 256, 0, stream>>>(x, Wg, eidx, wgt, cnt, partials);
    k_offs<<<1, 32, 0, stream>>>(cnt, offs);
    k_reduce<<<1, 64, 0, stream>>>(partials, out + (size_t)NTOK * CDIM);
    k_padinit<<<RMAX / 256, 256, 0, stream>>>(rowmap, roww);
    k_route<<<NTOK / 256, 256, 0, stream>>>(eidx, wgt, fill, offs, rowmap, roww);

    k_gemm1<<<dim3(NTILES, HDIM / 64), 256, 0, stream>>>(xb, W1t, W3t, rowmap, offs, hbuf);
    k_gemm2<<<dim3(NTILES, CDIM / 128), 256, 0, stream>>>(hbuf, W2t, rowmap, roww, offs, out);
}
